// RobustRnn_25237227832067
// MI455X (gfx1250) — compile-verified, matrix operand order checked by reading, not run on hardware
//
#include <hip/hip_runtime.h>
#include <hip/hip_bf16.h>

// ---------------------------------------------------------------------------
// RobustRnn on MI455X (gfx1250): persistent WMMA scan kernel.
//   B=64, NU=64, NX=NW=128, NY=32, T=2048
// Scan is latency-bound (2047 serialized steps); all weights live in LDS
// (f16, ~180KB of the WGP's 320KB), u is software-pipelined (prefetch t+1
// while computing t), and the output pass is fused into the scan so the
// 64MB states tensor never exists.
// ---------------------------------------------------------------------------

#define Bc  64
#define NUc 64
#define NXc 128
#define NWc 128
#define NYc 32
#define Tc  2048

typedef __attribute__((ext_vector_type(16))) _Float16 v16h;
typedef __attribute__((ext_vector_type(8)))  _Float16 v8h;
typedef __attribute__((ext_vector_type(4)))  _Float16 v4h;
typedef __attribute__((ext_vector_type(8)))  float    v8f;

// f16 weight pack offsets (in halves), all matrices row-major as in reference
#define OFF_CV   0        // (NW,NX) = diag(1/lam) @ C2tild
#define OFF_DV   16384    // (NW,NU) = diag(1/lam) @ Dtild
#define OFF_FW   24576    // (NX,NX)
#define OFF_B1   40960    // (NX,NW)
#define OFF_B2   57344    // (NX,NU)
#define OFF_EI   65536    // (NX,NX) = inv(E)
#define OFF_C1   81920    // (NY,NX)
#define OFF_D11  86016    // (NY,NW)
#define OFF_D12  90112    // (NY,NU)
#define WPACK_HALVES 92160

// workspace byte offsets
#define WS_WPACK 0                       // 184320 B f16 weights
#define WS_BIAS  184320                  // 288 floats: bv[128], F_b[128], by[32]
#define WS_AUG   185856                  // 128x256 fp32 Gauss-Jordan scratch
#define WS_U16   317440                  // (B,T,NU) f16 transposed inputs (16 MB)

// dynamic LDS: 288 floats bias, then wpack + h + w + eh + 2x u (halves)
#define SMEM_BYTES (1152 + (WPACK_HALVES + 2048 + 2048 + 2048 + 2048) * 2)

__device__ __forceinline__ v8f wmma_f16(v16h a, v16h b, v8f c) {
  return __builtin_amdgcn_wmma_f32_16x16x32_f16(
      /*neg_a=*/false, a, /*neg_b=*/false, b,
      /*c_mod=*/(short)0, c, /*reuse_a=*/false, /*reuse_b=*/false);
}

// Load a 16x32 f16 tile from LDS in the WMMA A layout (also used for B when
// the weight is stored as B^T row-major): lane ln = matrix row, K split as
// [hi*8 .. hi*8+7] in VGPRs 0-3 and [16+hi*8 ..] in VGPRs 4-7 (ISA 7.12.2).
__device__ __forceinline__ v16h ld_tile(const _Float16* base, int ld, int row,
                                        int col, int hi, int ln) {
  const _Float16* p = base + (row + ln) * ld + col + hi * 8;
  v8h lo = *(const v8h*)p;
  v8h hh = *(const v8h*)(p + 16);
  v16h r;
#pragma unroll
  for (int i = 0; i < 8; ++i) { r[i] = lo[i]; r[8 + i] = hh[i]; }
  return r;
}

// ---------------------------------------------------------------------------
// Prep 1: Gauss-Jordan inverse of E (128x128). E = I + 0.01*N -> no pivoting.
// ---------------------------------------------------------------------------
__global__ void gj_inverse_kernel(const float* __restrict__ E,
                                  float* __restrict__ aug,
                                  _Float16* __restrict__ einv_out) {
  __shared__ float rowk[256];
  __shared__ float colk[128];
  __shared__ float inv_piv;
  const int tid = threadIdx.x;

  for (int i = tid; i < 128 * 256; i += 256) {
    int r = i >> 8, c = i & 255;
    aug[i] = (c < 128) ? E[r * 128 + c] : ((c - 128 == r) ? 1.0f : 0.0f);
  }
  __syncthreads();

  for (int k = 0; k < 128; ++k) {
    if (tid == 0) inv_piv = 1.0f / aug[k * 256 + k];
    __syncthreads();
    float rk = aug[k * 256 + tid] * inv_piv;   // scale pivot row
    aug[k * 256 + tid] = rk;
    rowk[tid] = rk;
    if (tid < 128) colk[tid] = (tid == k) ? 0.0f : aug[tid * 256 + k];
    __syncthreads();
    for (int i = 0; i < 128; ++i) {            // eliminate column k
      float f = colk[i];
      if (f != 0.0f) aug[i * 256 + tid] -= f * rk;
    }
    __syncthreads();
  }
  for (int i = tid; i < 128 * 128; i += 256) {
    int r = i >> 7, c = i & 127;
    einv_out[i] = (_Float16)aug[r * 256 + 128 + c];
  }
}

// ---------------------------------------------------------------------------
// Prep 2: fold 1/lam into Cv/Dv, convert all weights to f16, copy biases.
// ---------------------------------------------------------------------------
__global__ void prep_weights_kernel(const float* __restrict__ Fw, const float* __restrict__ Fb,
                                    const float* __restrict__ B1, const float* __restrict__ B2,
                                    const float* __restrict__ C2, const float* __restrict__ Dt,
                                    const float* __restrict__ bv, const float* __restrict__ lam,
                                    const float* __restrict__ C1, const float* __restrict__ D11,
                                    const float* __restrict__ D12, const float* __restrict__ by,
                                    _Float16* __restrict__ wpack, float* __restrict__ bias) {
  const int i = blockIdx.x * blockDim.x + threadIdx.x;  // 0..16383
  if (i < 16384) {
    wpack[OFF_CV + i] = (_Float16)(C2[i] / lam[i >> 7]);
    wpack[OFF_FW + i] = (_Float16)Fw[i];
    wpack[OFF_B1 + i] = (_Float16)B1[i];
  }
  if (i < 8192) {
    wpack[OFF_DV + i] = (_Float16)(Dt[i] / lam[i >> 6]);
    wpack[OFF_B2 + i] = (_Float16)B2[i];
  }
  if (i < 4096) {
    wpack[OFF_C1  + i] = (_Float16)C1[i];
    wpack[OFF_D11 + i] = (_Float16)D11[i];
  }
  if (i < 2048) wpack[OFF_D12 + i] = (_Float16)D12[i];
  if (i < 128) { bias[i] = bv[i]; bias[128 + i] = Fb[i]; }
  if (i < 32)  bias[256 + i] = by[i];
}

// ---------------------------------------------------------------------------
// Prep 3: transpose u (B,NU,T) fp32 -> (B,T,NU) f16 so per-step tiles are
// contiguous 128-byte rows.
// ---------------------------------------------------------------------------
__global__ void prep_u_kernel(const float* __restrict__ u, _Float16* __restrict__ u16) {
  const int i = blockIdx.x * blockDim.x + threadIdx.x;
  if (i >= Bc * Tc * NUc) return;
  const int j = i & (NUc - 1);
  const int r = i / NUc;
  const int t = r & (Tc - 1);
  const int b = r / Tc;
  u16[i] = (_Float16)u[(b * NUc + j) * Tc + t];
}

// ---------------------------------------------------------------------------
// Persistent scan. 4 WGs x 16 batch rows, 8 wave32 waves each. Per step:
//   stage V : v = h@Cv^T + u@Dv^T + bv ; w = relu(v)            (6 WMMA/wave)
//   stage EH: eh = h@F^T + w@B1^T + u@B2^T + Fb                 (10 WMMA/wave)
//             + fused output y_t on waves 0-1                   (10 WMMA/wave)
//   stage H': h <- eh@Einv^T                                    (4 WMMA/wave)
// u_{t+1} is prefetched into registers at the top of step t and committed to
// the alternate LDS buffer after EH — the global round-trip is fully hidden
// behind ~16 chained WMMAs, and only 3 barriers/step remain.
// ---------------------------------------------------------------------------
__launch_bounds__(256, 1)
__global__ void scan_kernel(const _Float16* __restrict__ g_wpack,
                            const float* __restrict__ g_bias,
                            const _Float16* __restrict__ g_u16,
                            float* __restrict__ out) {
  extern __shared__ char smem[];
  float*    s_bias  = (float*)smem;                 // 288 floats
  _Float16* s_wpack = (_Float16*)(smem + 1152);     // 92160 halves
  _Float16* s_hbuf  = s_wpack + WPACK_HALVES;       // 16x128
  _Float16* s_wbuf  = s_hbuf + 2048;                // 16x128
  _Float16* s_ebuf  = s_wbuf + 2048;                // 16x128
  _Float16* s_ubuf  = s_ebuf + 2048;                // 2 x (16x64)

  const int tid   = threadIdx.x;
  const int wave  = tid >> 5;
  const int lane  = tid & 31;
  const int hi    = lane >> 4;
  const int ln    = lane & 15;
  const int bbase = blockIdx.x * 16;

  // one-time cooperative load of weights + biases, h0 = 0
  {
    const v8h* src = (const v8h*)g_wpack;
    v8h*       dst = (v8h*)s_wpack;
    for (int i = tid; i < WPACK_HALVES / 8; i += 256) dst[i] = src[i];
    for (int i = tid; i < 288; i += 256) s_bias[i] = g_bias[i];
    for (int i = tid; i < 2048; i += 256) s_hbuf[i] = (_Float16)0.0f;
  }

  const int uidx = tid * 4;        // each thread owns 4 halves of the u tile
  const int um   = uidx >> 6;
  const int uj   = uidx & 63;
  const _Float16* gu = g_u16 + (size_t)(bbase + um) * Tc * NUc + uj;

  // preload u_0 into buffer 0
  *(v4h*)(s_ubuf + uidx) = *(const v4h*)(gu);
  __syncthreads();

  int buf = 0;
  v4h ureg;
  for (int t = 0; t < Tc; ++t) {
    const _Float16* ub = s_ubuf + buf * 1024;

    // prefetch u_{t+1}; the wait lands at the ds_store after stage EH,
    // hiding the HBM round-trip behind the V+EH WMMA chains.
    const bool more = (t + 1 < Tc);
    if (more) ureg = *(const v4h*)(gu + (size_t)(t + 1) * NUc);

    // ---- stage V: one 16x16 tile of w per wave ----
    {
      const int n0 = wave * 16;
      v8f acc = {};
#pragma unroll
      for (int kb = 0; kb < NXc; kb += 32)
        acc = wmma_f16(ld_tile(s_hbuf, NXc, 0, kb, hi, ln),
                       ld_tile(s_wpack + OFF_CV, NXc, n0, kb, hi, ln), acc);
#pragma unroll
      for (int kb = 0; kb < NUc; kb += 32)
        acc = wmma_f16(ld_tile(ub, NUc, 0, kb, hi, ln),
                       ld_tile(s_wpack + OFF_DV, NUc, n0, kb, hi, ln), acc);
      const float bvv = s_bias[n0 + ln];
#pragma unroll
      for (int r = 0; r < 8; ++r) {
        float v = acc[r] + bvv;
        s_wbuf[(r + 8 * hi) * NWc + n0 + ln] = (_Float16)(v > 0.0f ? v : 0.0f);
      }
    }
    __syncthreads();   // wbuf ready

    // ---- stage EH: one 16x16 tile of eh per wave ----
    {
      const int c0 = wave * 16;
      v8f acc = {};
#pragma unroll
      for (int kb = 0; kb < NXc; kb += 32)
        acc = wmma_f16(ld_tile(s_hbuf, NXc, 0, kb, hi, ln),
                       ld_tile(s_wpack + OFF_FW, NXc, c0, kb, hi, ln), acc);
#pragma unroll
      for (int kb = 0; kb < NWc; kb += 32)
        acc = wmma_f16(ld_tile(s_wbuf, NWc, 0, kb, hi, ln),
                       ld_tile(s_wpack + OFF_B1, NWc, c0, kb, hi, ln), acc);
#pragma unroll
      for (int kb = 0; kb < NUc; kb += 32)
        acc = wmma_f16(ld_tile(ub, NUc, 0, kb, hi, ln),
                       ld_tile(s_wpack + OFF_B2, NUc, c0, kb, hi, ln), acc);
      const float fbv = s_bias[128 + c0 + ln];
#pragma unroll
      for (int r = 0; r < 8; ++r)
        s_ebuf[(r + 8 * hi) * NXc + c0 + ln] = (_Float16)(acc[r] + fbv);
    }

    // ---- fused output y_t (uses h_t, w_t, u_t) on waves 0-1 ----
    if (wave < 2) {
      const int y0 = wave * 16;
      v8f acc = {};
#pragma unroll
      for (int kb = 0; kb < NXc; kb += 32)
        acc = wmma_f16(ld_tile(s_hbuf, NXc, 0, kb, hi, ln),
                       ld_tile(s_wpack + OFF_C1, NXc, y0, kb, hi, ln), acc);
#pragma unroll
      for (int kb = 0; kb < NWc; kb += 32)
        acc = wmma_f16(ld_tile(s_wbuf, NWc, 0, kb, hi, ln),
                       ld_tile(s_wpack + OFF_D11, NWc, y0, kb, hi, ln), acc);
#pragma unroll
      for (int kb = 0; kb < NUc; kb += 32)
        acc = wmma_f16(ld_tile(ub, NUc, 0, kb, hi, ln),
                       ld_tile(s_wpack + OFF_D12, NUc, y0, kb, hi, ln), acc);
      const float byv = s_bias[256 + y0 + ln];
#pragma unroll
      for (int r = 0; r < 8; ++r)
        out[((size_t)(bbase + r + 8 * hi) * NYc + y0 + ln) * Tc + t] =
            acc[r] + byv;
    }
    if (t == Tc - 1) break;        // h_{T} not needed

    // commit prefetched u_{t+1} into the alternate buffer (visible after S3)
    if (more) *(v4h*)(s_ubuf + (buf ^ 1) * 1024 + uidx) = ureg;
    __syncthreads();   // ebuf ready, all hbuf/ubuf readers done

    // ---- stage H': h_{t+1} = eh @ Einv^T ----
    {
      const int c0 = wave * 16;
      v8f acc = {};
#pragma unroll
      for (int kb = 0; kb < NXc; kb += 32)
        acc = wmma_f16(ld_tile(s_ebuf, NXc, 0, kb, hi, ln),
                       ld_tile(s_wpack + OFF_EI, NXc, c0, kb, hi, ln), acc);
#pragma unroll
      for (int r = 0; r < 8; ++r)
        s_hbuf[(r + 8 * hi) * NXc + c0 + ln] = (_Float16)acc[r];
    }
    __syncthreads();   // hbuf ready for next step
    buf ^= 1;
  }
}

// ---------------------------------------------------------------------------
extern "C" void kernel_launch(void* const* d_in, const int* in_sizes, int n_in,
                              void* d_out, int out_size, void* d_ws, size_t ws_size,
                              hipStream_t stream) {
  (void)in_sizes; (void)n_in; (void)out_size; (void)ws_size;

  const float* g_u   = (const float*)d_in[0];
  const float* g_E   = (const float*)d_in[1];
  const float* g_Fw  = (const float*)d_in[2];
  const float* g_Fb  = (const float*)d_in[3];
  const float* g_B1  = (const float*)d_in[4];
  const float* g_B2  = (const float*)d_in[5];
  const float* g_C2  = (const float*)d_in[6];
  const float* g_Dt  = (const float*)d_in[7];
  const float* g_bv  = (const float*)d_in[8];
  const float* g_lam = (const float*)d_in[9];
  const float* g_C1  = (const float*)d_in[10];
  const float* g_D11 = (const float*)d_in[11];
  const float* g_D12 = (const float*)d_in[12];
  const float* g_by  = (const float*)d_in[13];

  char* ws = (char*)d_ws;
  _Float16* w_wpack = (_Float16*)(ws + WS_WPACK);
  float*    w_bias  = (float*)(ws + WS_BIAS);
  float*    w_aug   = (float*)(ws + WS_AUG);
  _Float16* w_u16   = (_Float16*)(ws + WS_U16);

  prep_weights_kernel<<<64, 256, 0, stream>>>(g_Fw, g_Fb, g_B1, g_B2, g_C2, g_Dt,
                                              g_bv, g_lam, g_C1, g_D11, g_D12, g_by,
                                              w_wpack, w_bias);
  gj_inverse_kernel<<<1, 256, 0, stream>>>(g_E, w_aug, w_wpack + OFF_EI);
  prep_u_kernel<<<(Bc * Tc * NUc + 255) / 256, 256, 0, stream>>>(g_u, w_u16);

  scan_kernel<<<dim3(Bc / 16), dim3(256), SMEM_BYTES, stream>>>(
      w_wpack, w_bias, w_u16, (float*)d_out);
}
